// GAT_50714973831350
// MI455X (gfx1250) — matrix operand order repaired and checked by reading, not compiled, not run on hardware
//
#include <hip/hip_runtime.h>
#include <hip/hip_bf16.h>
#include <math.h>

typedef __attribute__((ext_vector_type(16))) _Float16 v16h;
typedef __attribute__((ext_vector_type(8)))  _Float16 v8h;
typedef __attribute__((ext_vector_type(4)))  _Float16 v4h;
typedef __attribute__((ext_vector_type(8)))  float    v8f;

#define F_IN 512
#define HID  256
#define H1   8
#define C1   32
#define NCLS 10
#define NEG_SLOPE 0.2f

// ---- ordered-uint mapping so atomicMax(u32) implements float max ----
__device__ __forceinline__ unsigned mapf(float f) {
  unsigned b = __float_as_uint(f);
  return (b >> 31) ? ~b : (b | 0x80000000u);
}
__device__ __forceinline__ float unmapf(unsigned m) {
  unsigned b = (m >> 31) ? (m & 0x7FFFFFFFu) : ~m;
  return __uint_as_float(b);
}

__device__ __forceinline__ void edge_sd(const int* __restrict__ ei, int e, int E,
                                        int& s, int& d) {
  if (e < E) { s = ei[e]; d = ei[E + e]; }
  else       { s = e - E; d = e - E; }
}

// load 16 f16 as two 8-half chunks `gap` halves apart (16B-aligned b128 loads)
__device__ __forceinline__ v16h load_frag(const _Float16* __restrict__ p, int gap) {
  v8h lo = *(const v8h*)(p);
  v8h hi = *(const v8h*)(p + gap);
  v16h r;
#pragma unroll
  for (int i = 0; i < 8; ++i) { r[i] = lo[i]; r[8 + i] = hi[i]; }
  return r;
}

// =====================  one-time f32 -> f16 conversions  =====================
__global__ void conv_xh(const float* __restrict__ x, _Float16* __restrict__ xh,
                        int n4) {  // n4 = count of float4 groups
  int i = blockIdx.x * blockDim.x + threadIdx.x;
  if (i >= n4) return;
  float4 v = ((const float4*)x)[i];
  v4h h;
  h[0] = (_Float16)v.x; h[1] = (_Float16)v.y;
  h[2] = (_Float16)v.z; h[3] = (_Float16)v.w;
  ((v4h*)xh)[i] = h;
}

__global__ void conv_w1t(const float* __restrict__ W1, _Float16* __restrict__ w1t) {
  int t = blockIdx.x * blockDim.x + threadIdx.x;   // over 512*256
  if (t >= F_IN * HID) return;
  int n = t & (HID - 1);
  int k = t >> 8;
  w1t[(size_t)n * F_IN + k] = (_Float16)W1[t];     // read coalesced over n
}

// ==========  GEMM1: hsrc = x @ W1 via WMMA f16, direct global fragments  =====
// wave tile: 16(M) x 64(N); software-pipelined: k+1 fragments loaded before
// the four WMMAs of step k so loadcnt waits are partial, not 0x0.
__global__ __launch_bounds__(256) void gemm1_wmma(const _Float16* __restrict__ xh,
                                                  const _Float16* __restrict__ w1t,
                                                  float* __restrict__ hsrc,
                                                  int Nn, int Mtiles) {
  const int lane = threadIdx.x & 31;
  const int gw   = blockIdx.x * 8 + (threadIdx.x >> 5);
  const int m_tile  = gw >> 2;     // consecutive waves share the M tile
  const int n_strip = gw & 3;      // 4 strips of 64 columns
  if (m_tile >= Mtiles) return;

  const int row_l = lane & 15;
  const int hi    = lane >> 4;
  const int n0    = n_strip * 64;

  const int arow  = m_tile * 16 + row_l;
  const int arowc = arow < Nn ? arow : 0;              // clamp (store is guarded)
  const _Float16* abase = xh + (size_t)arowc * F_IN + hi * 8;

  const _Float16* bbase[4];
#pragma unroll
  for (int ni = 0; ni < 4; ++ni)
    bbase[ni] = w1t + (size_t)(n0 + ni * 16 + row_l) * F_IN + hi * 8;

  v8f acc[4] = {};

  // prologue: fragments for k-step 0
  v16h a_cur = load_frag(abase, 16);
  v16h b_cur[4];
#pragma unroll
  for (int ni = 0; ni < 4; ++ni) b_cur[ni] = load_frag(bbase[ni], 16);

#pragma unroll
  for (int kk = 0; kk < F_IN / 32; ++kk) {
    v16h a_nxt, b_nxt[4];
    if (kk + 1 < F_IN / 32) {
      const int k0n = (kk + 1) * 32;
      __builtin_prefetch(abase + k0n + 32, 0, 3);
      a_nxt = load_frag(abase + k0n, 16);
#pragma unroll
      for (int ni = 0; ni < 4; ++ni)
        b_nxt[ni] = load_frag(bbase[ni] + k0n, 16);
    }
#pragma unroll
    for (int ni = 0; ni < 4; ++ni)
      acc[ni] = __builtin_amdgcn_wmma_f32_16x16x32_f16(
          false, a_cur, false, b_cur[ni], (short)0, acc[ni], false, false);
    if (kk + 1 < F_IN / 32) {
      a_cur = a_nxt;
#pragma unroll
      for (int ni = 0; ni < 4; ++ni) b_cur[ni] = b_nxt[ni];
    }
  }

  // D layout: VGPR i -> row = i + 8*hi, col = lane&15
#pragma unroll
  for (int ni = 0; ni < 4; ++ni) {
    const int col = n0 + ni * 16 + row_l;
#pragma unroll
    for (int i = 0; i < 8; ++i) {
      const int row = m_tile * 16 + hi * 8 + i;
      if (row < Nn) hsrc[(size_t)row * HID + col] = acc[ni][i];
    }
  }
}

// =====================  attention scalars layer 1  =====================
__global__ void attn_dots1(const float* __restrict__ hsrc,
                           const float* __restrict__ att_s,
                           const float* __restrict__ att_d,
                           float* __restrict__ a_src, float* __restrict__ a_dst,
                           int Nn) {
  int idx = blockIdx.x * blockDim.x + threadIdx.x;
  if (idx >= Nn * H1) return;
  const int n = idx >> 3, hh = idx & 7;
  const float* hp = hsrc + (size_t)n * HID + hh * C1;
  const float* as = att_s + hh * C1;
  const float* ad = att_d + hh * C1;
  float ss = 0.f, sd = 0.f;
#pragma unroll
  for (int c = 0; c < C1; ++c) { ss += hp[c] * as[c]; sd += hp[c] * ad[c]; }
  a_src[idx] = ss;
  a_dst[idx] = sd;
}

// =====================  edge softmax passes (templated on heads)  ============
template <int H>
__global__ void edge_maxk(const int* __restrict__ ei, int E, int Nn,
                          const float* __restrict__ asrc,
                          const float* __restrict__ adst,
                          unsigned* __restrict__ emax) {
  int idx = blockIdx.x * blockDim.x + threadIdx.x;
  const int Etot = E + Nn;
  if (idx >= Etot * H) return;
  const int e = idx / H, hh = idx % H;
  int s, d; edge_sd(ei, e, E, s, d);
  float ev = asrc[s * H + hh] + adst[d * H + hh];
  ev = ev > 0.f ? ev : NEG_SLOPE * ev;
  atomicMax(&emax[d * H + hh], mapf(ev));
}

template <int H>
__global__ void edge_denomk(const int* __restrict__ ei, int E, int Nn,
                            const float* __restrict__ asrc,
                            const float* __restrict__ adst,
                            const unsigned* __restrict__ emax,
                            float* __restrict__ den) {
  int idx = blockIdx.x * blockDim.x + threadIdx.x;
  const int Etot = E + Nn;
  if (idx >= Etot * H) return;
  const int e = idx / H, hh = idx % H;
  int s, d; edge_sd(ei, e, E, s, d);
  float ev = asrc[s * H + hh] + adst[d * H + hh];
  ev = ev > 0.f ? ev : NEG_SLOPE * ev;
  atomicAdd(&den[d * H + hh], __expf(ev - unmapf(emax[d * H + hh])));
}

// =====================  layer-1 aggregation: one wave per edge  ==============
__global__ __launch_bounds__(256) void edge_aggr1(
    const int* __restrict__ ei, int E, int Nn,
    const float* __restrict__ asrc, const float* __restrict__ adst,
    const unsigned* __restrict__ emax, const float* __restrict__ den,
    const float* __restrict__ hsrc, float* __restrict__ agg) {
  const int eid  = blockIdx.x * 8 + (threadIdx.x >> 5);
  const int lane = threadIdx.x & 31;
  if (eid >= E + Nn) return;
  int s, d; edge_sd(ei, eid, E, s, d);
  const int hh = lane >> 2;  // 8 channels/lane -> head = (lane*8)/32
  float ev = asrc[s * H1 + hh] + adst[d * H1 + hh];
  ev = ev > 0.f ? ev : NEG_SLOPE * ev;
  const float al = __expf(ev - unmapf(emax[d * H1 + hh])) / den[d * H1 + hh];
  const int c0 = lane * 8;
  const float4* hp = (const float4*)(hsrc + (size_t)s * HID + c0);
  float4 v0 = hp[0], v1 = hp[1];
  float* op = agg + (size_t)d * HID + c0;
  atomicAdd(op + 0, al * v0.x); atomicAdd(op + 1, al * v0.y);
  atomicAdd(op + 2, al * v0.z); atomicAdd(op + 3, al * v0.w);
  atomicAdd(op + 4, al * v1.x); atomicAdd(op + 5, al * v1.y);
  atomicAdd(op + 6, al * v1.z); atomicAdd(op + 7, al * v1.w);
}

// =====================  bias + ELU (in place)  =====================
__global__ void elu_bias(float* __restrict__ agg, const float* __restrict__ b1,
                         int Nn) {
  int idx = blockIdx.x * blockDim.x + threadIdx.x;
  if (idx >= Nn * HID) return;
  float v = agg[idx] + b1[idx & (HID - 1)];
  agg[idx] = v > 0.f ? v : expm1f(v);
}

// =====================  GEMM2 + attention scalars: one wave per row  =========
__global__ __launch_bounds__(256) void gemm2_attn(
    const float* __restrict__ h2, const float* __restrict__ W2,
    const float* __restrict__ as2, const float* __restrict__ ad2,
    float* __restrict__ g2, float* __restrict__ a_src2,
    float* __restrict__ a_dst2, int Nn) {
  __shared__ float W2s[HID * NCLS];
  for (int i = threadIdx.x; i < HID * NCLS; i += 256) W2s[i] = W2[i];
  __syncthreads();
  const int n = blockIdx.x * 8 + (threadIdx.x >> 5);
  if (n >= Nn) return;
  const int lane = threadIdx.x & 31;
  float acc[NCLS];
#pragma unroll
  for (int c = 0; c < NCLS; ++c) acc[c] = 0.f;
  const float* hp = h2 + (size_t)n * HID;
  for (int k = lane; k < HID; k += 32) {
    const float xv = hp[k];
    const float* wr = &W2s[k * NCLS];
#pragma unroll
    for (int c = 0; c < NCLS; ++c) acc[c] += xv * wr[c];
  }
#pragma unroll
  for (int c = 0; c < NCLS; ++c)
    for (int off = 16; off; off >>= 1) acc[c] += __shfl_down(acc[c], off, 32);
  if (lane == 0) {
    float ss = 0.f, sd = 0.f;
#pragma unroll
    for (int c = 0; c < NCLS; ++c) {
      g2[(size_t)n * NCLS + c] = acc[c];
      ss += acc[c] * as2[c];
      sd += acc[c] * ad2[c];
    }
    a_src2[n] = ss;
    a_dst2[n] = sd;
  }
}

// =====================  layer-2 aggregation: one thread per edge  ============
__global__ void edge_aggr2(const int* __restrict__ ei, int E, int Nn,
                           const float* __restrict__ asrc,
                           const float* __restrict__ adst,
                           const unsigned* __restrict__ emax,
                           const float* __restrict__ den,
                           const float* __restrict__ g2,
                           float* __restrict__ out) {
  int e = blockIdx.x * blockDim.x + threadIdx.x;
  if (e >= E + Nn) return;
  int s, d; edge_sd(ei, e, E, s, d);
  float ev = asrc[s] + adst[d];
  ev = ev > 0.f ? ev : NEG_SLOPE * ev;
  const float al = __expf(ev - unmapf(emax[d])) / den[d];
#pragma unroll
  for (int c = 0; c < NCLS; ++c)
    atomicAdd(&out[(size_t)d * NCLS + c], al * g2[(size_t)s * NCLS + c]);
}

// =====================  bias + log_softmax (in place on d_out)  ==============
__global__ void final_lsm(float* __restrict__ out, const float* __restrict__ b2,
                          int Nn) {
  int n = blockIdx.x * blockDim.x + threadIdx.x;
  if (n >= Nn) return;
  float v[NCLS];
  float mx = -3.402823466e38f;
#pragma unroll
  for (int c = 0; c < NCLS; ++c) {
    v[c] = out[(size_t)n * NCLS + c] + b2[c];
    mx = fmaxf(mx, v[c]);
  }
  float se = 0.f;
#pragma unroll
  for (int c = 0; c < NCLS; ++c) se += __expf(v[c] - mx);
  const float ls = logf(se);
#pragma unroll
  for (int c = 0; c < NCLS; ++c) out[(size_t)n * NCLS + c] = v[c] - mx - ls;
}

// ============================================================================
extern "C" void kernel_launch(void* const* d_in, const int* in_sizes, int n_in,
                              void* d_out, int out_size, void* d_ws, size_t ws_size,
                              hipStream_t stream) {
  const float* x   = (const float*)d_in[0];
  const int*   ei  = (const int*)d_in[1];
  const float* W1  = (const float*)d_in[2];
  const float* as1 = (const float*)d_in[3];
  const float* ad1 = (const float*)d_in[4];
  const float* b1  = (const float*)d_in[5];
  const float* W2  = (const float*)d_in[6];
  const float* as2 = (const float*)d_in[7];
  const float* ad2 = (const float*)d_in[8];
  const float* b2  = (const float*)d_in[9];
  float* out = (float*)d_out;

  const int Nn   = in_sizes[0] / F_IN;
  const int E    = in_sizes[1] / 2;
  const int Etot = E + Nn;
  const int Mtiles = (Nn + 15) / 16;

  // workspace carve-up
  char* ws = (char*)d_ws;
  size_t off = 0;
  auto alloc = [&](size_t bytes) -> void* {
    void* p = ws + off;
    off += (bytes + 255) & ~(size_t)255;
    return p;
  };
  _Float16* xh    = (_Float16*)alloc((size_t)Nn * F_IN * 2);
  _Float16* w1t   = (_Float16*)alloc((size_t)F_IN * HID * 2);
  float*    hsrc  = (float*)alloc((size_t)Nn * HID * 4);
  float*    agg1  = (float*)alloc((size_t)Nn * HID * 4);
  float*    asrc1 = (float*)alloc((size_t)Nn * H1 * 4);
  float*    adst1 = (float*)alloc((size_t)Nn * H1 * 4);
  unsigned* emax1 = (unsigned*)alloc((size_t)Nn * H1 * 4);
  float*    den1  = (float*)alloc((size_t)Nn * H1 * 4);
  float*    g2    = (float*)alloc((size_t)Nn * NCLS * 4);
  float*    asrc2 = (float*)alloc((size_t)Nn * 4);
  float*    adst2 = (float*)alloc((size_t)Nn * 4);
  unsigned* emax2 = (unsigned*)alloc((size_t)Nn * 4);
  float*    den2  = (float*)alloc((size_t)Nn * 4);
  (void)ws_size; (void)n_in; (void)out_size;

  hipMemsetAsync(agg1,  0, (size_t)Nn * HID * 4, stream);
  hipMemsetAsync(emax1, 0, (size_t)Nn * H1 * 4, stream);
  hipMemsetAsync(den1,  0, (size_t)Nn * H1 * 4, stream);
  hipMemsetAsync(emax2, 0, (size_t)Nn * 4, stream);
  hipMemsetAsync(den2,  0, (size_t)Nn * 4, stream);
  hipMemsetAsync(out,   0, (size_t)Nn * NCLS * 4, stream);

  // conversions
  const int n4 = Nn * (F_IN / 4);
  conv_xh<<<(n4 + 255) / 256, 256, 0, stream>>>(x, xh, n4);
  conv_w1t<<<(F_IN * HID + 255) / 256, 256, 0, stream>>>(W1, w1t);

  // layer 1
  const int nwaves = Mtiles * 4;
  gemm1_wmma<<<(nwaves + 7) / 8, 256, 0, stream>>>(xh, w1t, hsrc, Nn, Mtiles);
  attn_dots1<<<(Nn * H1 + 255) / 256, 256, 0, stream>>>(hsrc, as1, ad1, asrc1, adst1, Nn);
  edge_maxk<H1><<<(Etot * H1 + 255) / 256, 256, 0, stream>>>(ei, E, Nn, asrc1, adst1, emax1);
  edge_denomk<H1><<<(Etot * H1 + 255) / 256, 256, 0, stream>>>(ei, E, Nn, asrc1, adst1, emax1, den1);
  edge_aggr1<<<(Etot + 7) / 8, 256, 0, stream>>>(ei, E, Nn, asrc1, adst1, emax1, den1, hsrc, agg1);
  elu_bias<<<(Nn * HID + 255) / 256, 256, 0, stream>>>(agg1, b1, Nn);

  // layer 2
  gemm2_attn<<<(Nn + 7) / 8, 256, 0, stream>>>(agg1, W2, as2, ad2, g2, asrc2, adst2, Nn);
  edge_maxk<1><<<(Etot + 255) / 256, 256, 0, stream>>>(ei, E, Nn, asrc2, adst2, emax2);
  edge_denomk<1><<<(Etot + 255) / 256, 256, 0, stream>>>(ei, E, Nn, asrc2, adst2, emax2, den2);
  edge_aggr2<<<(Etot + 255) / 256, 256, 0, stream>>>(ei, E, Nn, asrc2, adst2, emax2, den2, g2, out);
  final_lsm<<<(Nn + 255) / 256, 256, 0, stream>>>(out, b2, Nn);
}
